// SA_33028298506668
// MI455X (gfx1250) — matrix operand order repaired and checked by reading, not compiled, not run on hardware
//
#include <hip/hip_runtime.h>

// ---------------------------------------------------------------------------
// MI455X (gfx1250) fused self-attention block.
// bf16 WMMA (V_WMMA_F32_16X16X32_BF16) with fp32 accumulation everywhere;
// flash-attention style fusion so the [B,H,S,S] score tensor never hits HBM.
// Global->LDS staging uses the Tensor Data Mover (TENSOR_LOAD_TO_LDS,
// TENSORcnt double-buffering) when the builtin is available.
// ---------------------------------------------------------------------------

typedef __bf16 bf16;
typedef __attribute__((ext_vector_type(16))) __bf16 v16bf;
typedef __attribute__((ext_vector_type(8)))  __bf16 v8bf;
typedef __attribute__((ext_vector_type(8)))  float  v8f;
typedef __attribute__((ext_vector_type(4)))  unsigned int u32x4;
typedef __attribute__((ext_vector_type(8)))  int i32x8;
typedef __attribute__((ext_vector_type(4)))  int i32x4;

#define B_NUM 8
#define S_LEN 1024
#define D_DIM 1024
#define H_NUM 16
#define HD    64
#define NTOK  (B_NUM * S_LEN)   // 8192

#define BKDIM 64
#define LDT   72                // LDS stride (bf16) for 64-wide tiles  (144B, 16B-mult)
#define LDP   136               // LDS stride (bf16) for 128-wide tiles (272B, 16B-mult)

#if __has_builtin(__builtin_amdgcn_tensor_load_to_lds)
#define HAVE_TDM 1
#else
#define HAVE_TDM 0
#endif

__device__ __forceinline__ v8f zero8() {
  v8f z = {0.f,0.f,0.f,0.f,0.f,0.f,0.f,0.f};
  return z;
}

__device__ __forceinline__ v8f wmma_bf(v16bf a, v16bf b, v8f c) {
  // D = A(16x32 bf16) * B(32x16 bf16) + C(16x16 f32)
  return __builtin_amdgcn_wmma_f32_16x16x32_bf16(
      /*neg_a=*/false, a, /*neg_b=*/false, b,
      /*c_mod=*/(short)0, c, /*reuse_a=*/false, /*reuse_b=*/false);
}

__device__ __forceinline__ v16bf pack16(v8bf lo, v8bf hi) {
  v16bf r;
#pragma unroll
  for (int j = 0; j < 8; ++j) { r[j] = lo[j]; r[j + 8] = hi[j]; }
  return r;
}

// A-fragment (16x32 bf16, M=lane&15): elements 0..7 at K=8*kh, 8..15 at K=16+8*kh
__device__ __forceinline__ v16bf load_frag_a(const bf16* p) {
  return pack16(*(const v8bf*)p, *(const v8bf*)(p + 16));
}
// B-fragment (32x16 bf16, N=lane&15): elements 0..7 at K=8*kh, 8..15 at K=16+8*kh
__device__ __forceinline__ v16bf load_frag_b(const bf16* p) {
  return pack16(*(const v8bf*)p, *(const v8bf*)(p + 16));
}

__device__ __forceinline__ float hmax16(float v) {
  v = fmaxf(v, __shfl_xor(v, 1, 32));
  v = fmaxf(v, __shfl_xor(v, 2, 32));
  v = fmaxf(v, __shfl_xor(v, 4, 32));
  v = fmaxf(v, __shfl_xor(v, 8, 32));
  return v;
}
__device__ __forceinline__ float hsum16(float v) {
  v += __shfl_xor(v, 1, 32);
  v += __shfl_xor(v, 2, 32);
  v += __shfl_xor(v, 4, 32);
  v += __shfl_xor(v, 8, 32);
  return v;
}

#if HAVE_TDM
// low 32 bits of a generic pointer to __shared__ == LDS byte offset
__device__ __forceinline__ unsigned lds_off(const void* p) {
  return (unsigned)(size_t)p;
}

// TDM 2D tile load: tile_w elements per row (bf16), tile_h rows, source row
// stride = stride_elems.  LDS destination gets 16B padding after every 128B
// row (pad_interval code 4 = 32 DWORDs, pad_amount code 3 = 4 DWORDs), which
// reproduces the LDT=72 layout used by the WMMA fragment loaders.
__device__ __forceinline__ void tdm_load_tile_2d(
    unsigned lds_byte_off, const void* gsrc,
    unsigned tile_w, unsigned tile_h, unsigned stride_elems)
{
  unsigned long long ga = (unsigned long long)(size_t)gsrc;
  u32x4 g0;
  g0[0] = 1u;                                        // count=1 (valid user D#)
  g0[1] = lds_byte_off;                              // lds_addr
  g0[2] = (unsigned)(ga & 0xFFFFFFFFull);            // global_addr[31:0]
  g0[3] = (unsigned)((ga >> 32) & 0x1FFFFFFull)      // global_addr[56:32]
        | (2u << 30);                                // type=2 ("image")
  i32x8 g1;
  g1[0] = (int)((1u << 16)        // data_size=1 -> 2 bytes
              | (1u << 20)        // pad_enable
              | (4u << 22)        // pad_interval: 32 DWORDs (128B)
              | (3u << 25));      // pad_amount: 4 DWORDs (16B)
  g1[1] = (int)((tile_w & 0xFFFFu) << 16);                   // tensor_dim0 lo16
  g1[2] = (int)(((tile_w >> 16) & 0xFFFFu)
              | ((tile_h & 0xFFFFu) << 16));                 // dim0 hi | dim1 lo
  g1[3] = (int)(((tile_h >> 16) & 0xFFFFu)
              | ((tile_w & 0xFFFFu) << 16));                 // dim1 hi | tile_dim0
  g1[4] = (int)(tile_h & 0xFFFFu);                           // tile_dim1 (tile_dim2=0)
  g1[5] = (int)stride_elems;                                 // tensor_dim0_stride lo32
  g1[6] = 0;
  g1[7] = 0;
  i32x4 z4 = {0, 0, 0, 0};
#if __clang_major__ >= 23
  i32x8 z8 = {0, 0, 0, 0, 0, 0, 0, 0};
  __builtin_amdgcn_tensor_load_to_lds(g0, g1, z4, z4, z8, 0);
#else
  __builtin_amdgcn_tensor_load_to_lds(g0, g1, z4, z4, 0);
#endif
}
#endif  // HAVE_TDM

// ---------------------------------------------------------------------------
// fp32 -> bf16 conversion
// ---------------------------------------------------------------------------
__global__ void cvt_kernel(const float* __restrict__ src, bf16* __restrict__ dst, int n) {
  int i = blockIdx.x * blockDim.x + threadIdx.x;
  int stride = gridDim.x * blockDim.x;
  for (; i < n; i += stride) dst[i] = (bf16)src[i];
}

// ---------------------------------------------------------------------------
// 128x128 GEMM tile mainloop: C[row,col] = sum_k A[row,k] * W[col,k]
// (i.e. x @ W^T, torch Linear).  A, W are bf16 row-major with K stride D_DIM.
// 8 waves: wave (wr,wc) computes 32x64 (2x4 tiles of 16x16).
// NBUF LDS buffers: TDM double-buffers, fallback uses one buffer.
// ---------------------------------------------------------------------------
#if HAVE_TDM
#define NBUF 2
#else
#define NBUF 1
#endif

__device__ __forceinline__ void gemm_compute_slab(
    const bf16* __restrict__ cA, const bf16* __restrict__ cB, v8f acc[2][4])
{
  const int lane = threadIdx.x & 31, wave = threadIdx.x >> 5;
  const int kh = lane >> 4, cl = lane & 15;
  const int wr = wave >> 1, wc = wave & 1;
#pragma unroll
  for (int ks = 0; ks < 2; ++ks) {
    v16bf af[2];
#pragma unroll
    for (int rt = 0; rt < 2; ++rt)
      af[rt] = load_frag_a(cA + (wr * 32 + rt * 16 + cl) * LDT + ks * 32 + 8 * kh);
#pragma unroll
    for (int ct = 0; ct < 4; ++ct) {
      v16bf bf = load_frag_b(cB + (wc * 64 + ct * 16 + cl) * LDT + ks * 32 + 8 * kh);
      acc[0][ct] = wmma_bf(af[0], bf, acc[0][ct]);
      acc[1][ct] = wmma_bf(af[1], bf, acc[1][ct]);
    }
  }
}

__device__ __forceinline__ void gemm_tile_128x128(
    const bf16* __restrict__ A, const bf16* __restrict__ W,
    int rowbase, int colbase,
    bf16 (*sA)[128 * LDT], bf16 (*sB)[128 * LDT], v8f acc[2][4])
{
  const int tid = threadIdx.x;
  const int wave = tid >> 5;
  const int NK = D_DIM / BKDIM;

#if HAVE_TDM
  const bf16* arow = A + (size_t)rowbase * D_DIM;
  const bf16* brow = W + (size_t)colbase * D_DIM;
  if (wave == 0) {
    tdm_load_tile_2d(lds_off(sA[0]), arow, BKDIM, 128, D_DIM);
    tdm_load_tile_2d(lds_off(sB[0]), brow, BKDIM, 128, D_DIM);
  }
#pragma unroll 1
  for (int kt = 0; kt < NK; ++kt) {
    const int cur = kt & 1;
    if (wave == 0) {
      if (kt + 1 < NK) {
        // kick next slab into the other buffer, then wait for current slab
        tdm_load_tile_2d(lds_off(sA[cur ^ 1]), arow + (kt + 1) * BKDIM, BKDIM, 128, D_DIM);
        tdm_load_tile_2d(lds_off(sB[cur ^ 1]), brow + (kt + 1) * BKDIM, BKDIM, 128, D_DIM);
        __builtin_amdgcn_s_wait_tensorcnt(2);   // oldest 2 (current slab) done
      } else {
        __builtin_amdgcn_s_wait_tensorcnt(0);
      }
    }
    __syncthreads();
    gemm_compute_slab(sA[cur], sB[cur], acc);
    __syncthreads();
  }
#else
  const int r = tid >> 1, cs = (tid & 1) * 32;
  const bf16* ag = A + (size_t)(rowbase + r) * D_DIM + cs;
  const bf16* bg = W + (size_t)(colbase + r) * D_DIM + cs;
  bf16* sa = sA[0] + r * LDT + cs;
  bf16* sb = sB[0] + r * LDT + cs;
#pragma unroll 1
  for (int kt = 0; kt < NK; ++kt) {
    const bf16* agk = ag + kt * BKDIM;
    const bf16* bgk = bg + kt * BKDIM;
#pragma unroll
    for (int j = 0; j < 4; ++j) {
      *(v8bf*)(sa + j * 8) = *(const v8bf*)(agk + j * 8);
      *(v8bf*)(sb + j * 8) = *(const v8bf*)(bgk + j * 8);
    }
    if (kt + 1 < NK) {
      __builtin_prefetch(agk + BKDIM, 0, 3);
      __builtin_prefetch(bgk + BKDIM, 0, 3);
    }
    __syncthreads();
    gemm_compute_slab(sA[0], sB[0], acc);
    __syncthreads();
  }
#endif
  (void)wave;
}

// ---------------------------------------------------------------------------
// QKV projection: z selects {Wq,bq->q}, {Wk,bk->k}, {Wv,bv->v};
// output written bf16 in heads layout [B,H,S,HD].
// ---------------------------------------------------------------------------
__global__ __launch_bounds__(256) void gemm_qkv_kernel(
    const bf16* __restrict__ ybf,
    const bf16* __restrict__ wq, const bf16* __restrict__ wk, const bf16* __restrict__ wv,
    const float* __restrict__ bq, const float* __restrict__ bk, const float* __restrict__ bv,
    bf16* __restrict__ qo, bf16* __restrict__ ko, bf16* __restrict__ vo)
{
  __shared__ __align__(16) bf16 sA[NBUF][128 * LDT];
  __shared__ __align__(16) bf16 sB[NBUF][128 * LDT];

  const bf16* W; const float* bias; bf16* out;
  if      (blockIdx.z == 0) { W = wq; bias = bq; out = qo; }
  else if (blockIdx.z == 1) { W = wk; bias = bk; out = ko; }
  else                      { W = wv; bias = bv; out = vo; }

  const int rowbase = blockIdx.x * 128, colbase = blockIdx.y * 128;
  v8f acc[2][4];
#pragma unroll
  for (int a = 0; a < 2; ++a)
#pragma unroll
    for (int c = 0; c < 4; ++c) acc[a][c] = zero8();

  gemm_tile_128x128(ybf, W, rowbase, colbase, sA, sB, acc);

  const int lane = threadIdx.x & 31, wave = threadIdx.x >> 5;
  const int kh = lane >> 4, cl = lane & 15;
  const int wr = wave >> 1, wc = wave & 1;
#pragma unroll
  for (int rt = 0; rt < 2; ++rt)
#pragma unroll
    for (int ct = 0; ct < 4; ++ct) {
      int colf = colbase + wc * 64 + ct * 16 + cl;
      float bia = bias[colf];
      int hh = colf >> 6, hd = colf & 63;
#pragma unroll
      for (int i = 0; i < 8; ++i) {
        int row = rowbase + wr * 32 + rt * 16 + i + 8 * kh;   // token index
        int bb = row >> 10, ss = row & 1023;
        // [B,H,S,HD]: S*HD == 65536
        out[(((size_t)(bb * H_NUM + hh)) << 16) + (ss << 6) + hd] =
            (bf16)(acc[rt][ct][i] + bia);
      }
    }
}

// ---------------------------------------------------------------------------
// Flash attention: block = (q-tile of 128, one (b,h)); 8 waves x 16 q-rows.
// ---------------------------------------------------------------------------
__global__ __launch_bounds__(256) void attn_kernel(
    const bf16* __restrict__ qb, const bf16* __restrict__ kbuf,
    const bf16* __restrict__ vbuf, const unsigned char* __restrict__ maskp,
    bf16* __restrict__ ab)
{
  __shared__ __align__(16) bf16 sK [128 * LDT];       // K tile  [key][hd]
  __shared__ __align__(16) bf16 sVt[ 64 * LDP];       // V tile transposed [hd][key]
  __shared__ __align__(16) bf16 sP [8 * 16 * LDP];    // per-wave P staging
  __shared__ unsigned char sM[128];

  const int tid = threadIdx.x, lane = tid & 31, wave = tid >> 5;
  const int kh = lane >> 4, cl = lane & 15;
  const int bh = blockIdx.y, b = bh >> 4, h = bh & 15;
  const int qbase = blockIdx.x * 128 + wave * 16;

  // Q strip fragments for this wave: 16 rows x 64 hd -> 2 K-steps
  const bf16* qrow = qb + ((size_t)bh * S_LEN + qbase + cl) * HD;
  v16bf qf[2];
#pragma unroll
  for (int s = 0; s < 2; ++s)
    qf[s] = load_frag_a(qrow + s * 32 + 8 * kh);

  float mrun[8], lrun[8];
#pragma unroll
  for (int i = 0; i < 8; ++i) { mrun[i] = -1e30f; lrun[i] = 0.f; }
  v8f o[4];
#pragma unroll
  for (int t = 0; t < 4; ++t) o[t] = zero8();

  bf16* sPw = sP + wave * 16 * LDP;
  const int r = tid >> 1, cs = (tid & 1) * 32;
  const bf16* vg = vbuf + ((size_t)bh * S_LEN + r) * HD + cs;
#if !HAVE_TDM
  const bf16* kg = kbuf + ((size_t)bh * S_LEN + r) * HD + cs;
#endif

  for (int blk = 0; blk < S_LEN / 128; ++blk) {
    // ---- stage K (TDM DMA when available) and V (transposed) into LDS ----
#if HAVE_TDM
    if (wave == 0)
      tdm_load_tile_2d(lds_off(sK), kbuf + ((size_t)bh * S_LEN + blk * 128) * HD,
                       HD, 128, HD);
#else
    const bf16* kgk = kg + (size_t)blk * 128 * HD;
#pragma unroll
    for (int j = 0; j < 4; ++j)
      *(v8bf*)(sK + r * LDT + cs + j * 8) = *(const v8bf*)(kgk + j * 8);
#endif
    const bf16* vgk = vg + (size_t)blk * 128 * HD;
#pragma unroll
    for (int j = 0; j < 4; ++j) {
      v8bf vv = *(const v8bf*)(vgk + j * 8);
#pragma unroll
      for (int e = 0; e < 8; ++e)
        sVt[(cs + j * 8 + e) * LDP + r] = vv[e];
    }
    if (tid < 128) sM[tid] = maskp[b * S_LEN + blk * 128 + tid];  // jax bool = 1 byte
#if HAVE_TDM
    if (wave == 0) __builtin_amdgcn_s_wait_tensorcnt(0);
#endif
    __syncthreads();

    // ---- S = Q K^T / sqrt(HD), masked ----
    v8f sf[8];
#pragma unroll
    for (int t = 0; t < 8; ++t) {
      const bf16* p0 = sK + (t * 16 + cl) * LDT + 8 * kh;        // B frag: B[k][key]=K[key][k]
      v8f c = wmma_bf(qf[0], load_frag_b(p0), zero8());
      c = wmma_bf(qf[1], load_frag_b(p0 + 32), c);
      bool msk = sM[t * 16 + cl] != 0;
#pragma unroll
      for (int i = 0; i < 8; ++i)
        c[i] = msk ? -1e9f : (c[i] * 0.125f);
      sf[t] = c;
    }

    // ---- online softmax ----
    float corr[8];
#pragma unroll
    for (int i = 0; i < 8; ++i) {
      float v = sf[0][i];
#pragma unroll
      for (int t = 1; t < 8; ++t) v = fmaxf(v, sf[t][i]);
      v = hmax16(v);
      float mn = fmaxf(mrun[i], v);
      corr[i] = __expf(mrun[i] - mn);
      mrun[i] = mn;
    }
    float psum[8];
#pragma unroll
    for (int i = 0; i < 8; ++i) psum[i] = 0.f;
#pragma unroll
    for (int t = 0; t < 8; ++t)
#pragma unroll
      for (int i = 0; i < 8; ++i) {
        float p = __expf(sf[t][i] - mrun[i]);
        psum[i] += p;
        sPw[(i + 8 * kh) * LDP + t * 16 + cl] = (bf16)p;   // D-frag -> row-major P
      }
#pragma unroll
    for (int i = 0; i < 8; ++i)
      lrun[i] = lrun[i] * corr[i] + hsum16(psum[i]);
#pragma unroll
    for (int t = 0; t < 4; ++t)
#pragma unroll
      for (int i = 0; i < 8; ++i) o[t][i] *= corr[i];

    // ---- O += P V  (P reloaded as A-fragments; same-wave LDS is in-order) ----
#pragma unroll
    for (int s = 0; s < 4; ++s) {
      v16bf af = load_frag_a(sPw + cl * LDP + s * 32 + 8 * kh);
#pragma unroll
      for (int t = 0; t < 4; ++t) {
        v16bf bfg = load_frag_b(sVt + (t * 16 + cl) * LDP + s * 32 + 8 * kh);
        o[t] = wmma_bf(af, bfg, o[t]);
      }
    }
    __syncthreads();
  }

  // ---- write normalized O back to token-major [N,D] bf16 ----
#pragma unroll
  for (int t = 0; t < 4; ++t)
#pragma unroll
    for (int i = 0; i < 8; ++i) {
      int srow = qbase + i + 8 * kh;
      int feat = h * 64 + t * 16 + cl;
      ab[((size_t)b * S_LEN + srow) * D_DIM + feat] = (bf16)(o[t][i] / lrun[i]);
    }
}

// ---------------------------------------------------------------------------
// Output projection: proj = atted @ Wm^T + bm, fp32 into d_out (temp)
// ---------------------------------------------------------------------------
__global__ __launch_bounds__(256) void gemm_proj_kernel(
    const bf16* __restrict__ abf, const bf16* __restrict__ wm,
    const float* __restrict__ bm, float* __restrict__ outf)
{
  __shared__ __align__(16) bf16 sA[NBUF][128 * LDT];
  __shared__ __align__(16) bf16 sB[NBUF][128 * LDT];

  const int rowbase = blockIdx.x * 128, colbase = blockIdx.y * 128;
  v8f acc[2][4];
#pragma unroll
  for (int a = 0; a < 2; ++a)
#pragma unroll
    for (int c = 0; c < 4; ++c) acc[a][c] = zero8();

  gemm_tile_128x128(abf, wm, rowbase, colbase, sA, sB, acc);

  const int lane = threadIdx.x & 31, wave = threadIdx.x >> 5;
  const int kh = lane >> 4, cl = lane & 15;
  const int wr = wave >> 1, wc = wave & 1;
#pragma unroll
  for (int rt = 0; rt < 2; ++rt)
#pragma unroll
    for (int ct = 0; ct < 4; ++ct) {
      int colf = colbase + wc * 64 + ct * 16 + cl;
      float bia = bm[colf];
#pragma unroll
      for (int i = 0; i < 8; ++i) {
        int row = rowbase + wr * 32 + rt * 16 + i + 8 * kh;
        outf[(size_t)row * D_DIM + colf] = acc[rt][ct][i] + bia;
      }
    }
}

// ---------------------------------------------------------------------------
// Residual + LayerNorm (torch: unbiased std, eps added to std). In-place on out.
// ---------------------------------------------------------------------------
__global__ __launch_bounds__(256) void ln_kernel(
    const float* __restrict__ y, const float* __restrict__ a2,
    const float* __restrict__ b2, float* __restrict__ out)
{
  __shared__ float red[256];
  const int row = blockIdx.x, tid = threadIdx.x;
  const float* yr = y + (size_t)row * D_DIM;
  float* orow = out + (size_t)row * D_DIM;

  float x[4], s = 0.f;
#pragma unroll
  for (int j = 0; j < 4; ++j) {
    int c = tid + j * 256;
    x[j] = yr[c] + orow[c];      // residual
    s += x[j];
  }
  red[tid] = s; __syncthreads();
  for (int off = 128; off > 0; off >>= 1) {
    if (tid < off) red[tid] += red[tid + off];
    __syncthreads();
  }
  float mean = red[0] * (1.0f / 1024.0f);
  __syncthreads();

  float ss = 0.f;
#pragma unroll
  for (int j = 0; j < 4; ++j) { float d = x[j] - mean; ss += d * d; }
  red[tid] = ss; __syncthreads();
  for (int off = 128; off > 0; off >>= 1) {
    if (tid < off) red[tid] += red[tid + off];
    __syncthreads();
  }
  float stdv = sqrtf(red[0] * (1.0f / 1023.0f));   // ddof=1
  float inv = 1.0f / (stdv + 1e-6f);               // eps on std
#pragma unroll
  for (int j = 0; j < 4; ++j) {
    int c = tid + j * 256;
    orow[c] = a2[c] * (x[j] - mean) * inv + b2[c];
  }
}

// ---------------------------------------------------------------------------
extern "C" void kernel_launch(void* const* d_in, const int* in_sizes, int n_in,
                              void* d_out, int out_size, void* d_ws, size_t ws_size,
                              hipStream_t stream) {
  const float* y  = (const float*)d_in[0];
  const unsigned char* ymask = (const unsigned char*)d_in[1]; // bool, 1B/elem
  const float* Wq = (const float*)d_in[2];
  const float* bq = (const float*)d_in[3];
  const float* Wk = (const float*)d_in[4];
  const float* bk = (const float*)d_in[5];
  const float* Wv = (const float*)d_in[6];
  const float* bv = (const float*)d_in[7];
  const float* Wm = (const float*)d_in[8];
  const float* bm = (const float*)d_in[9];
  const float* a2 = (const float*)d_in[10];
  const float* b2 = (const float*)d_in[11];
  float* out = (float*)d_out;

  // workspace layout (bf16 elements); total ~88 MB
  bf16* ws  = (bf16*)d_ws;
  size_t off = 0;
  bf16* ybf = ws + off; off += (size_t)NTOK * D_DIM;
  bf16* wqb = ws + off; off += (size_t)D_DIM * D_DIM;
  bf16* wkb = ws + off; off += (size_t)D_DIM * D_DIM;
  bf16* wvb = ws + off; off += (size_t)D_DIM * D_DIM;
  bf16* wmb = ws + off; off += (size_t)D_DIM * D_DIM;
  bf16* qb  = ws + off; off += (size_t)NTOK * D_DIM;   // [B,H,S,HD]
  bf16* kb  = ws + off; off += (size_t)NTOK * D_DIM;
  bf16* vb  = ws + off; off += (size_t)NTOK * D_DIM;
  bf16* abf = ws + off; off += (size_t)NTOK * D_DIM;   // atted [N,D]

  cvt_kernel<<<2048, 256, 0, stream>>>(y,  ybf, NTOK * D_DIM);
  cvt_kernel<<<1024, 256, 0, stream>>>(Wq, wqb, D_DIM * D_DIM);
  cvt_kernel<<<1024, 256, 0, stream>>>(Wk, wkb, D_DIM * D_DIM);
  cvt_kernel<<<1024, 256, 0, stream>>>(Wv, wvb, D_DIM * D_DIM);
  cvt_kernel<<<1024, 256, 0, stream>>>(Wm, wmb, D_DIM * D_DIM);

  gemm_qkv_kernel<<<dim3(NTOK / 128, D_DIM / 128, 3), 256, 0, stream>>>(
      ybf, wqb, wkb, wvb, bq, bk, bv, qb, kb, vb);

  attn_kernel<<<dim3(S_LEN / 128, B_NUM * H_NUM), 256, 0, stream>>>(
      qb, kb, vb, ymask, abf);

  gemm_proj_kernel<<<dim3(NTOK / 128, D_DIM / 128), 256, 0, stream>>>(
      abf, wmb, bm, out);

  ln_kernel<<<NTOK, 256, 0, stream>>>(y, a2, b2, out);
}
